// CausalSelfAttention_17497696764253
// MI455X (gfx1250) — matrix operand order repaired and checked by reading, not compiled, not run on hardware
//
#include <hip/hip_runtime.h>
#include <math.h>
#include <stdint.h>

// ---------------- problem constants ----------------
#define D_MODEL   1024
#define NUM_HEADS 16
#define HEAD_DIM  64
#define BATCH     2
#define SEQ       2048
#define M_TOTAL   (BATCH * SEQ)   // 4096 rows through the projections

// ---------------- types ----------------
typedef __attribute__((ext_vector_type(16))) __bf16        v16bf;
typedef __attribute__((ext_vector_type(8)))  float         v8f;
typedef __attribute__((ext_vector_type(4)))  unsigned int  u32x4;
typedef __attribute__((ext_vector_type(4)))  float         f32x4;
typedef __attribute__((ext_vector_type(2)))  unsigned int  u32x2;

union FragB {            // 16 bf16 operand fragment == 8 VGPRs
    v16bf v;
    u32x4 u[2];
};

// fp32 -> bf16, round-to-nearest-even
__device__ inline unsigned short f2bf(float f) {
    unsigned u = __float_as_uint(f);
    u += 0x7FFFu + ((u >> 16) & 1u);
    return (unsigned short)(u >> 16);
}

__device__ inline v8f wmma_bf16(const FragB& a, const FragB& b, v8f c) {
    return __builtin_amdgcn_wmma_f32_16x16x32_bf16(false, a.v, false, b.v,
                                                   (short)0, c, false, false);
}

// CDNA5 async copy: global -> LDS, 16 bytes per lane, tracked by ASYNCcnt
__device__ inline void async_g2l_b128(unsigned lds_addr, const void* gaddr) {
    asm volatile("global_load_async_to_lds_b128 %0, %1, off"
                 :: "v"(lds_addr), "v"((unsigned long long)gaddr) : "memory");
}
__device__ inline void wait_async0() {
    asm volatile("s_wait_asynccnt 0x0" ::: "memory");
}

// 16-lane (half-wave) reductions; xor of bits 0..3 stays inside each half of wave32
__device__ inline float rmax16(float v) {
#pragma unroll
    for (int m = 1; m < 16; m <<= 1) v = fmaxf(v, __shfl_xor(v, m, 32));
    return v;
}
__device__ inline float rsum16(float v) {
#pragma unroll
    for (int m = 1; m < 16; m <<= 1) v += __shfl_xor(v, m, 32);
    return v;
}

// ---------------- kernel 0: fp32 -> bf16 convert (4 elems/thread) ----------------
__global__ void cvt_f32_bf16(const float* __restrict__ in,
                             unsigned short* __restrict__ out, int n4) {
    int i = blockIdx.x * blockDim.x + threadIdx.x;
    int stride = gridDim.x * blockDim.x;
    for (; i < n4; i += stride) {
        f32x4 f = *(const f32x4*)(in + (size_t)i * 4);
        u32x2 p;
        p.x = (unsigned)f2bf(f.x) | ((unsigned)f2bf(f.y) << 16);
        p.y = (unsigned)f2bf(f.z) | ((unsigned)f2bf(f.w) << 16);
        *(u32x2*)(out + (size_t)i * 4) = p;
    }
}

// ---------------- kernel 1: WMMA GEMM  Y[m,n] = sum_k A[m,k]*B[n,k] + bias[n] ----
// A: bf16 [M,K] row-major.  Bw: bf16 [N,K] row-major (torch Linear W).
// B tile (64x32 per k-step) staged block-wide through LDS with CDNA5 async
// copies; k-loop manually unrolled 2x with ping-pong register/buffer sets
// (compile-time indices -> no v_mov double-buffer swaps).
// Each wave owns a 32x64 tile: 2 A-frags x 4 B-frags = 8 WMMA per k-step.
// mode 0: bf16 out [B,H,S,Dh]; mode 1: bf16 out [B,H,Dh,S]; mode 2: fp32 [M,D_MODEL]
// Grid: (N/64, M/256), block 256 = 8 waves.  Requires K % 64 == 0.
__global__ void __launch_bounds__(256)
gemm_bf16_nt(const unsigned short* __restrict__ A,
             const unsigned short* __restrict__ Bw,
             const float* __restrict__ bias,
             void* __restrict__ out, int K, int mode) {
    __shared__ __align__(16) unsigned short btile[2][64][32];   // 2 x 4KB

    const int lane = threadIdx.x & 31;
    const int wave = threadIdx.x >> 5;
    const int lo = lane & 15, hi = lane >> 4;
    const int m0 = blockIdx.y * 256 + wave * 32;
    const int n0 = blockIdx.x * 64;

    // async staging assignment: wave w copies B-tile rows [8w, 8w+8), 16B per lane
    const int srow = 8 * wave + (lane >> 2);
    const int sseg = (lane & 3) * 8;                       // element offset in row
    const unsigned short* sgp = Bw + (size_t)(n0 + srow) * K + sseg;

    // prologue: stage B(k=0) into buffer 0, load A(k=0) fragments (rows m0, m0+16)
    async_g2l_b128((unsigned)(uintptr_t)&btile[0][srow][sseg], sgp);
    const unsigned short* ap0 = A + (size_t)(m0 + lo) * K + 16 * hi;
    const unsigned short* ap1 = ap0 + (size_t)16 * K;
    FragB aF[2][2];                       // [buffer][m-subtile]
    aF[0][0].u[0] = *(const u32x4*)(ap0);
    aF[0][0].u[1] = *(const u32x4*)(ap0 + 8);
    aF[0][1].u[0] = *(const u32x4*)(ap1);
    aF[0][1].u[1] = *(const u32x4*)(ap1 + 8);
    wait_async0();

    v8f acc[2][4] = {};

    // One pipeline stage: compute k-step from {btile,aF}[CUR], prefetch k=KNEXT
    // into {btile,aF}[NXT].  CUR/NXT are literal constants.
#define GEMM_STAGE(CUR, NXT, KNEXT)                                           \
    {                                                                         \
        __syncthreads();  /* btile[CUR] written; btile[NXT] readers done */   \
        if ((KNEXT) < K) {                                                    \
            async_g2l_b128((unsigned)(uintptr_t)&btile[NXT][srow][sseg],      \
                           sgp + (KNEXT));                                    \
            aF[NXT][0].u[0] = *(const u32x4*)(ap0 + (KNEXT));                 \
            aF[NXT][0].u[1] = *(const u32x4*)(ap0 + (KNEXT) + 8);             \
            aF[NXT][1].u[0] = *(const u32x4*)(ap1 + (KNEXT));                 \
            aF[NXT][1].u[1] = *(const u32x4*)(ap1 + (KNEXT) + 8);             \
        }                                                                     \
        _Pragma("unroll")                                                     \
        for (int nt = 0; nt < 4; ++nt) {                                      \
            FragB bfr;                                                        \
            const unsigned short* bp = &btile[CUR][nt * 16 + lo][16 * hi];    \
            bfr.u[0] = *(const u32x4*)(bp);                                   \
            bfr.u[1] = *(const u32x4*)(bp + 8);                               \
            acc[0][nt] = wmma_bf16(aF[CUR][0], bfr, acc[0][nt]);              \
            acc[1][nt] = wmma_bf16(aF[CUR][1], bfr, acc[1][nt]);              \
        }                                                                     \
        wait_async0();  /* my writes to btile[NXT] done before next barrier */\
    }

    for (int k0 = 0; k0 < K; k0 += 64) {
        GEMM_STAGE(0, 1, k0 + 32)
        GEMM_STAGE(1, 0, k0 + 64)
    }
#undef GEMM_STAGE

    // C/D layout: VGPR r, lanes 0-15 -> M=r, lanes 16-31 -> M=8+r; N = lo
#pragma unroll
    for (int i = 0; i < 2; ++i) {
#pragma unroll
        for (int r = 0; r < 8; ++r) {
            const int m = m0 + 16 * i + r + 8 * hi;
            const int bt = m >> 11;          // m / SEQ
            const int s = m & (SEQ - 1);
#pragma unroll
            for (int nt = 0; nt < 4; ++nt) {
                const int n = n0 + nt * 16 + lo;
                const float val = acc[i][nt][r] + bias[n];
                if (mode == 2) {
                    ((float*)out)[(size_t)m * D_MODEL + n] = val;
                } else {
                    const int h = n >> 6, d = n & 63;
                    size_t idx;
                    if (mode == 0)
                        idx = (((size_t)(bt * NUM_HEADS + h) * SEQ + s) << 6) + d;
                    else
                        idx = ((size_t)(bt * NUM_HEADS + h) * HEAD_DIM + d) * SEQ + s;
                    ((unsigned short*)out)[idx] = f2bf(val);
                }
            }
        }
    }
}

// ---------------- kernel 2: flash attention (causal) ----------------
// Q,K: bf16 [B,H,S,64].  Vt: bf16 [B,H,64,S].  AO: bf16 [B,S,1024] (heads merged).
// Grid: (B*H, S/128), block 256 = 8 waves; each wave owns one 16-row q tile.
__global__ void __launch_bounds__(256)
attn_kernel(const unsigned short* __restrict__ Qb,
            const unsigned short* __restrict__ Kb,
            const unsigned short* __restrict__ Vtb,
            unsigned short* __restrict__ AO) {
    __shared__ __align__(16) unsigned short pbuf[8][16][32];   // per-wave P tile

    const int lane = threadIdx.x & 31;
    const int wave = threadIdx.x >> 5;
    const int lo = lane & 15, hi = lane >> 4;
    const int bh = blockIdx.x;
    const int b = bh >> 4, h = bh & 15;
    const int q0 = blockIdx.y * 128 + wave * 16;

    const unsigned short* Qh = Qb + (size_t)bh * SEQ * HEAD_DIM;
    const unsigned short* Kh = Kb + (size_t)bh * SEQ * HEAD_DIM;
    const unsigned short* Vh = Vtb + (size_t)bh * HEAD_DIM * SEQ;

    // Hoist Q fragments (16x64 tile -> two 16x32 A-fragments over d)
    FragB aQ[2];
#pragma unroll
    for (int dd = 0; dd < 2; ++dd) {
        const unsigned short* qp =
            Qh + (size_t)(q0 + lo) * HEAD_DIM + dd * 32 + 16 * hi;
        aQ[dd].u[0] = *(const u32x4*)qp;
        aQ[dd].u[1] = *(const u32x4*)(qp + 8);
    }

    float mrun[8], lrun[8];
#pragma unroll
    for (int r = 0; r < 8; ++r) { mrun[r] = -INFINITY; lrun[r] = 0.0f; }
    v8f o[4] = {};                       // 16x64 output tile (4 d-subtiles)

    const float scale = 0.125f;          // 1/sqrt(HEAD_DIM)
    const int kend = q0 + 16;            // causal bound (exclusive)

    for (int kc = 0; kc < kend; kc += 32) {
        // ---- scores S = Q K^T for keys [kc, kc+32) : two 16x16 accumulators ----
        v8f s0 = {}, s1 = {};
#pragma unroll
        for (int dd = 0; dd < 2; ++dd) {
            FragB b0, b1;
            const unsigned short* kp0 =
                Kh + (size_t)(kc + lo) * HEAD_DIM + dd * 32 + 16 * hi;
            b0.u[0] = *(const u32x4*)kp0;
            b0.u[1] = *(const u32x4*)(kp0 + 8);
            const unsigned short* kp1 =
                Kh + (size_t)(kc + 16 + lo) * HEAD_DIM + dd * 32 + 16 * hi;
            b1.u[0] = *(const u32x4*)kp1;
            b1.u[1] = *(const u32x4*)(kp1 + 8);
            s0 = wmma_bf16(aQ[dd], b0, s0);
            s1 = wmma_bf16(aQ[dd], b1, s1);
        }

        // prefetch V fragments now -- independent of softmax, hides load latency
        FragB bv[4];
#pragma unroll
        for (int nt = 0; nt < 4; ++nt) {
            const unsigned short* vp =
                Vh + (size_t)(nt * 16 + lo) * SEQ + kc + 16 * hi;
            bv[nt].u[0] = *(const u32x4*)vp;
            bv[nt].u[1] = *(const u32x4*)(vp + 8);
        }

        // ---- causal mask + online softmax (rows M = r + 8*hi, cols = lo) ----
#pragma unroll
        for (int r = 0; r < 8; ++r) {
            const int q = q0 + r + 8 * hi;
            float v0 = s0[r] * scale; if (kc + lo > q)      v0 = -INFINITY;
            float v1 = s1[r] * scale; if (kc + 16 + lo > q) v1 = -INFINITY;
            const float mx   = rmax16(fmaxf(v0, v1));
            const float mnew = fmaxf(mrun[r], mx);        // finite from 1st tile on
            const float corr = __expf(mrun[r] - mnew);    // 0 on first tile
            const float p0 = __expf(v0 - mnew);
            const float p1 = __expf(v1 - mnew);
            lrun[r] = lrun[r] * corr + rsum16(p0 + p1);
            mrun[r] = mnew;
#pragma unroll
            for (int nt = 0; nt < 4; ++nt) o[nt][r] *= corr;
            // stage P as bf16 in D-layout: row = r+8*hi, cols lo / 16+lo
            pbuf[wave][r + 8 * hi][lo]      = f2bf(p0);
            pbuf[wave][r + 8 * hi][16 + lo] = f2bf(p1);
        }
        // cross-lane LDS dependency is invisible to the compiler: order + wait
        asm volatile("s_wait_dscnt 0x0" ::: "memory");

        // re-read P in A-fragment layout (row = lo, k-base = 16*hi)
        FragB aP;
        const unsigned short* pp = &pbuf[wave][lo][16 * hi];
        aP.u[0] = *(const u32x4*)pp;
        aP.u[1] = *(const u32x4*)(pp + 8);
        asm volatile("" ::: "memory");

        // ---- O += P * V  (V fragments already in registers) ----
#pragma unroll
        for (int nt = 0; nt < 4; ++nt)
            o[nt] = wmma_bf16(aP, bv[nt], o[nt]);
    }

    // ---- normalize and write merged-head bf16 output [B,S,1024] ----
#pragma unroll
    for (int r = 0; r < 8; ++r) {
        const float inv = 1.0f / lrun[r];
        const int sq = q0 + r + 8 * hi;
#pragma unroll
        for (int nt = 0; nt < 4; ++nt) {
            const int d = nt * 16 + lo;
            AO[(((size_t)(b * SEQ + sq)) << 10) + h * 64 + d] =
                f2bf(o[nt][r] * inv);
        }
    }
}

// ---------------- host-side launch ----------------
extern "C" void kernel_launch(void* const* d_in, const int* in_sizes, int n_in,
                              void* d_out, int out_size, void* d_ws, size_t ws_size,
                              hipStream_t stream) {
    const float* x  = (const float*)d_in[0];
    const float* Wq = (const float*)d_in[1];
    const float* bq = (const float*)d_in[2];
    const float* Wk = (const float*)d_in[3];
    const float* bk = (const float*)d_in[4];
    const float* Wv = (const float*)d_in[5];
    const float* bv = (const float*)d_in[6];
    const float* Wo = (const float*)d_in[7];
    const float* bo = (const float*)d_in[8];
    float* out = (float*)d_out;

    // workspace partition (bf16 buffers), total = 48 MB
    const size_t NX = (size_t)M_TOTAL * D_MODEL;   // 4M elems
    const size_t NW = (size_t)D_MODEL * D_MODEL;   // 1M elems
    unsigned short* xb  = (unsigned short*)d_ws;
    unsigned short* wqb = xb  + NX;
    unsigned short* wkb = wqb + NW;
    unsigned short* wvb = wkb + NW;
    unsigned short* wob = wvb + NW;
    unsigned short* Qb  = wob + NW;      // [B,H,S,64]
    unsigned short* Kbf = Qb  + NX;      // [B,H,S,64]
    unsigned short* Vtb = Kbf + NX;      // [B,H,64,S]
    unsigned short* AO  = Vtb + NX;      // [B,S,1024]

    // 1) fp32 -> bf16 conversions (4 elems / thread)
    cvt_f32_bf16<<<1024, 256, 0, stream>>>(x,  xb,  (int)(NX / 4));
    cvt_f32_bf16<<<512,  256, 0, stream>>>(Wq, wqb, (int)(NW / 4));
    cvt_f32_bf16<<<512,  256, 0, stream>>>(Wk, wkb, (int)(NW / 4));
    cvt_f32_bf16<<<512,  256, 0, stream>>>(Wv, wvb, (int)(NW / 4));
    cvt_f32_bf16<<<512,  256, 0, stream>>>(Wo, wob, (int)(NW / 4));

    // 2) Q/K/V projections (V written pre-transposed)
    dim3 gg(D_MODEL / 64, M_TOTAL / 256);   // (16, 16)
    gemm_bf16_nt<<<gg, 256, 0, stream>>>(xb, wqb, bq, Qb,  D_MODEL, 0);
    gemm_bf16_nt<<<gg, 256, 0, stream>>>(xb, wkb, bk, Kbf, D_MODEL, 0);
    gemm_bf16_nt<<<gg, 256, 0, stream>>>(xb, wvb, bv, Vtb, D_MODEL, 1);

    // 3) causal flash attention -> merged-head bf16
    attn_kernel<<<dim3(BATCH * NUM_HEADS, SEQ / 128), 256, 0, stream>>>(Qb, Kbf, Vtb, AO);

    // 4) output projection (fp32 result with bias)
    gemm_bf16_nt<<<gg, 256, 0, stream>>>(AO, wob, bo, out, D_MODEL, 2);
}
